// DeformableAttention_75909251990032
// MI455X (gfx1250) — compile-verified, matrix operand order checked by reading, not run on hardware
//
#include <hip/hip_runtime.h>
#include <math.h>

typedef __attribute__((ext_vector_type(16))) _Float16 v16h;
typedef __attribute__((ext_vector_type(8)))  float    v8f;

#define NTOK    40000
#define GRIDHW  200
#define DMODEL  256
#define NHEADS  8
#define NPOINTS 8
#define DHEAD   32
#define KSTEPS  (DMODEL / 32)      // 8
#define MTILES  (NTOK / 16)        // 2500

// ---------------------------------------------------------------------------
// A-fragment-major layout (f16): frag[(mtile*KSTEPS + ks)*32 + lane][e 0..15]
//   lane = ((k>>3)&1)*16 + (m&15),  e = ((k>>4)<<3) | (k&7)   (ISA 7.12.2)
// Each lane's fragment is a 32-byte contiguous chunk -> one aligned load.
// ---------------------------------------------------------------------------
__global__ __launch_bounds__(256)
void convert_A_frag(const float* __restrict__ A, _Float16* __restrict__ Af)
{
    const int f = blockIdx.x * 256 + threadIdx.x;      // fragment id
    if (f >= MTILES * KSTEPS * 32) return;
    const int lane  = f & 31;
    const int ks    = (f >> 5) & (KSTEPS - 1);
    const int mtile = f >> 8;
    const int row   = mtile * 16 + (lane & 15);
    const int kbase = ks * 32 + ((lane >> 4) << 3);    // half*8
    const float* src = A + (size_t)row * DMODEL + kbase;
    _Float16 tmp[16];
#pragma unroll
    for (int e = 0; e < 8; ++e)  tmp[e]     = (_Float16)src[e];
#pragma unroll
    for (int e = 0; e < 8; ++e)  tmp[8 + e] = (_Float16)src[16 + e];
    *(v16h*)(Af + (size_t)f * 16) = *(const v16h*)tmp;
}

// ---------------------------------------------------------------------------
// B-fragment-major layout (f16): frag[((ks*(NCOLS/16) + ntile)*32 + lane)][e]
//   source W[k][n] row-major: k = ks*32 + (e>>3)*16 + (lane>>4)*8 + (e&7),
//                             n = ntile*16 + (lane&15)
// ---------------------------------------------------------------------------
template <int NCOLS>
__global__ __launch_bounds__(256)
void convert_B_frag(const float* __restrict__ W, _Float16* __restrict__ Bf)
{
    const int idx = blockIdx.x * 256 + threadIdx.x;    // element id
    if (idx >= DMODEL * NCOLS) return;
    const int e     = idx & 15;
    const int lane  = (idx >> 4) & 31;
    const int ntile = (idx >> 9) % (NCOLS / 16);
    const int ks    = idx / (512 * (NCOLS / 16));
    const int k = ks * 32 + ((e >> 3) << 4) + ((lane >> 4) << 3) + (e & 7);
    const int n = ntile * 16 + (lane & 15);
    Bf[idx] = (_Float16)W[(size_t)k * NCOLS + n];
}

// ---------------------------------------------------------------------------
// GEMM on pre-swizzled fragments: C = A @ W + bias.
// Block = 128 threads (4 waves); wave owns one 16-row tile x 64 cols.
// Inner loop: pure 32B global fragment loads + v_wmma (no LDS, no barriers).
// MODE 0: f32 row-major [NTOK][NCOLS]
// MODE 1: f16 head-major vmap [(col/32)][row][col%32]
// ---------------------------------------------------------------------------
template <int NCOLS, int MODE>
__global__ __launch_bounds__(128)
void gemm_frag(const _Float16* __restrict__ Af, const _Float16* __restrict__ Bf,
               const float* __restrict__ bias, void* __restrict__ out)
{
    const int lane   = threadIdx.x & 31;
    const int wv     = threadIdx.x >> 5;
    const int half   = lane >> 4;
    const int mtile  = blockIdx.x * 4 + wv;
    const int n0tile = blockIdx.y * 4;
    constexpr int NT = NCOLS / 16;

    const v16h* Ap = (const v16h*)Af + (size_t)mtile * KSTEPS * 32 + lane;
    const v16h* Bp = (const v16h*)Bf + (size_t)n0tile * 32 + lane;

    v8f acc[4] = {};
#pragma unroll
    for (int ks = 0; ks < KSTEPS; ++ks) {
        const v16h a = Ap[ks * 32];
#pragma unroll
        for (int t = 0; t < 4; ++t) {
            const v16h b = Bp[((size_t)ks * NT + t) * 32];
            acc[t] = __builtin_amdgcn_wmma_f32_16x16x32_f16(
                false, a, false, b, (short)0, acc[t], false, false);
        }
    }

    // D fragment (VGPR r, lane): M = mtile*16 + r + 8*half, N = 16*t + lane%16
#pragma unroll
    for (int t = 0; t < 4; ++t) {
        const int col = (n0tile + t) * 16 + (lane & 15);
        const float bc = bias[col];
#pragma unroll
        for (int r = 0; r < 8; ++r) {
            const int row = mtile * 16 + r + (half << 3);
            const float v = acc[t][r] + bc;
            if (MODE == 0) {
                ((float*)out)[(size_t)row * NCOLS + col] = v;
            } else {
                const int nh = col >> 5, dh = col & 31;
                ((_Float16*)out)[((size_t)nh * NTOK + row) * DHEAD + dh] =
                    (_Float16)v;
            }
        }
    }
}

// ---------------------------------------------------------------------------
// Deformable sampling: block per token, wave = head, lane = head channel.
// Scalar math is wave-uniform; corner gathers are 64B coalesced f16 reads
// from the L2-resident head-major value map. Output goes straight to
// A-fragment-major f16 so the output GEMM consumes it with no extra pass.
// ---------------------------------------------------------------------------
__global__ __launch_bounds__(256)
void deform_sample(const float* __restrict__ offs,    // NTOK x 128
                   const float* __restrict__ logits,  // NTOK x 64
                   const float* __restrict__ refpts,  // NTOK x 2
                   const _Float16* __restrict__ vmap, // NHEADS x NTOK x 32
                   _Float16* __restrict__ preF)       // A-frag-major f16
{
    const int n    = blockIdx.x;
    const int h    = threadIdx.x >> 5;
    const int lane = threadIdx.x & 31;

    const float* off = offs   + (size_t)n * 128 + h * 16;
    const float* lg  = logits + (size_t)n * 64  + h * 8;
    const float rx = refpts[2 * n + 0];
    const float ry = refpts[2 * n + 1];

    float w[NPOINTS];
    float mx = lg[0];
#pragma unroll
    for (int p = 1; p < NPOINTS; ++p) mx = fmaxf(mx, lg[p]);
    float s = 0.f;
#pragma unroll
    for (int p = 0; p < NPOINTS; ++p) { w[p] = __expf(lg[p] - mx); s += w[p]; }
    const float inv = 1.0f / s;

    const _Float16* vh = vmap + (size_t)h * NTOK * DHEAD;
    float acc = 0.f;

#pragma unroll
    for (int p = 0; p < NPOINTS; ++p) {
        const float ap = w[p] * inv;
        float lx = rx + off[2 * p + 0] * (1.0f / GRIDHW);
        float ly = ry + off[2 * p + 1] * (1.0f / GRIDHW);
        lx = fminf(fmaxf(lx, 0.f), 1.f);
        ly = fminf(fmaxf(ly, 0.f), 1.f);
        const float x = lx * (float)GRIDHW - 0.5f;   // align_corners=False
        const float y = ly * (float)GRIDHW - 0.5f;
        const float x0f = floorf(x), y0f = floorf(y);
        const float wx = x - x0f,   wy = y - y0f;
        const int x0 = (int)x0f, y0 = (int)y0f;
        const int x1 = x0 + 1,   y1 = y0 + 1;

        const float w00 = ap * (1.f - wx) * (1.f - wy);
        const float w01 = ap * wx * (1.f - wy);
        const float w10 = ap * (1.f - wx) * wy;
        const float w11 = ap * wx * wy;

        if (y0 >= 0 && x0 >= 0)
            acc += w00 * (float)vh[((size_t)(y0 * GRIDHW + x0)) * DHEAD + lane];
        if (y0 >= 0 && x1 < GRIDHW)
            acc += w01 * (float)vh[((size_t)(y0 * GRIDHW + x1)) * DHEAD + lane];
        if (y1 < GRIDHW && x0 >= 0)
            acc += w10 * (float)vh[((size_t)(y1 * GRIDHW + x0)) * DHEAD + lane];
        if (y1 < GRIDHW && x1 < GRIDHW)
            acc += w11 * (float)vh[((size_t)(y1 * GRIDHW + x1)) * DHEAD + lane];
    }

    // store into A-fragment-major f16 for the output GEMM:
    // row = n, k = h*32 + lane
    const int mtile = n >> 4, mm = n & 15;
    const int lfrag = (((lane >> 3) & 1) << 4) | mm;
    const int e     = ((lane >> 4) << 3) | (lane & 7);
    preF[(((size_t)mtile * KSTEPS + h) * 32 + lfrag) * 16 + e] = (_Float16)acc;
}

// ---------------------------------------------------------------------------
extern "C" void kernel_launch(void* const* d_in, const int* in_sizes, int n_in,
                              void* d_out, int out_size, void* d_ws, size_t ws_size,
                              hipStream_t stream)
{
    const float* query  = (const float*)d_in[0];
    const float* value  = (const float*)d_in[1];
    const float* refp   = (const float*)d_in[2];
    const float* W_off  = (const float*)d_in[3];
    const float* b_off  = (const float*)d_in[4];
    const float* W_attn = (const float*)d_in[5];
    const float* b_attn = (const float*)d_in[6];
    const float* W_val  = (const float*)d_in[7];
    const float* b_val  = (const float*)d_in[8];
    const float* W_out  = (const float*)d_in[9];
    const float* b_out  = (const float*)d_in[10];

    char* ws = (char*)d_ws;
    _Float16* qf      = (_Float16*)ws; ws += (size_t)NTOK * DMODEL * 2;
    _Float16* vf      = (_Float16*)ws; ws += (size_t)NTOK * DMODEL * 2;
    _Float16* bf_off  = (_Float16*)ws; ws += (size_t)DMODEL * 128 * 2;
    _Float16* bf_attn = (_Float16*)ws; ws += (size_t)DMODEL * 64 * 2;
    _Float16* bf_val  = (_Float16*)ws; ws += (size_t)DMODEL * DMODEL * 2;
    _Float16* bf_out  = (_Float16*)ws; ws += (size_t)DMODEL * DMODEL * 2;
    float*    d_offs  = (float*)ws;    ws += (size_t)NTOK * 128 * 4;
    float*    d_logit = (float*)ws;    ws += (size_t)NTOK * 64 * 4;
    _Float16* d_vproj = (_Float16*)ws; ws += (size_t)NTOK * DMODEL * 2;
    _Float16* d_preF  = (_Float16*)ws; ws += (size_t)NTOK * DMODEL * 2;

    const int nfragA = MTILES * KSTEPS * 32;                    // 640000
    convert_A_frag<<<dim3((nfragA + 255) / 256), dim3(256), 0, stream>>>(query, qf);
    convert_A_frag<<<dim3((nfragA + 255) / 256), dim3(256), 0, stream>>>(value, vf);
    convert_B_frag<128><<<dim3((DMODEL * 128 + 255) / 256), dim3(256), 0, stream>>>(W_off,  bf_off);
    convert_B_frag<64> <<<dim3((DMODEL * 64  + 255) / 256), dim3(256), 0, stream>>>(W_attn, bf_attn);
    convert_B_frag<256><<<dim3((DMODEL * 256 + 255) / 256), dim3(256), 0, stream>>>(W_val,  bf_val);
    convert_B_frag<256><<<dim3((DMODEL * 256 + 255) / 256), dim3(256), 0, stream>>>(W_out,  bf_out);

    const dim3 blk(128);
    gemm_frag<128, 0><<<dim3(625, 2), blk, 0, stream>>>(qf, bf_off,  b_off,  d_offs);
    gemm_frag<64,  0><<<dim3(625, 1), blk, 0, stream>>>(qf, bf_attn, b_attn, d_logit);
    gemm_frag<256, 1><<<dim3(625, 4), blk, 0, stream>>>(vf, bf_val,  b_val,  d_vproj);

    deform_sample<<<dim3(NTOK), dim3(256), 0, stream>>>(d_offs, d_logit, refp,
                                                        d_vproj, d_preF);

    gemm_frag<256, 0><<<dim3(625, 4), blk, 0, stream>>>(d_preF, bf_out, b_out, d_out);
}
